// CrossAttentionPlus_25615184953790
// MI455X (gfx1250) — compile-verified, matrix operand order checked by reading, not run on hardware
//
#include <hip/hip_runtime.h>
#include <hip/hip_bf16.h>

// ---------------------------------------------------------------------------
// CDNA5 (gfx1250) fused cross-attention:
//   wave32 + v_wmma_f32_16x16x32_f16 + global_load_async_to_lds_b128
// ---------------------------------------------------------------------------

typedef __attribute__((ext_vector_type(16))) _Float16 v16h;
typedef __attribute__((ext_vector_type(8)))  _Float16 v8h;
typedef __attribute__((ext_vector_type(8)))  float    v8f;

#define BB   4
#define QQ   1024
#define KK   1024
#define DIMM 1024
#define HH   16
#define DD   64
#define HD   1024           // H*D
#define SCALE 0.125f        // 1/sqrt(64)
#define NEGV (-1.0e9f)

static __device__ inline v8f wmma_f16(v16h a, v16h b, v8f c) {
    return __builtin_amdgcn_wmma_f32_16x16x32_f16(
        /*neg_a=*/false, a, /*neg_b=*/false, b,
        /*c_mod=*/(short)0, c, /*reuse_a=*/false, /*reuse_b=*/false);
}

static __device__ inline v16h combine8(v8h lo, v8h hi) {
    v16h r;
#pragma unroll
    for (int i = 0; i < 8; ++i) { r[i] = lo[i]; r[8 + i] = hi[i]; }
    return r;
}

// A/B fragment of a 16x32 f16 tile from row-major storage (stride in halves).
// Per ISA: lanes 0-15 row M=lane, K halves {k..k+7, k+16..k+23};
//          lanes 16-31 row M=lane-16, K halves {k+8..k+15, k+24..k+31}.
static __device__ inline v16h frag_ld(const _Float16* p, int stride,
                                      int rbase, int kbase) {
    int lane = threadIdx.x & 31;
    const _Float16* q = p + (size_t)(rbase + (lane & 15)) * stride
                          + kbase + ((lane >> 4) ? 8 : 0);
    v8h lo = *(const v8h*)(q);
    v8h hi = *(const v8h*)(q + 16);
    return combine8(lo, hi);
}

// Async global -> LDS copy (16B per lane), tracked by ASYNCcnt.
// GV mode: per-lane 64-bit global address; VDST = per-lane LDS byte offset.
static __device__ inline void async_copy_b128(unsigned lds_byte_off,
                                              const void* gptr) {
    asm volatile("global_load_async_to_lds_b128 %0, %1, off"
                 :: "v"(lds_byte_off),
                    "v"((unsigned long long)(uintptr_t)gptr)
                 : "memory");
}
static __device__ inline void wait_async0() {
    asm volatile("s_wait_asynccnt 0x0" ::: "memory");
}

// ---------------------------------------------------------------------------
// fp32 -> f16 conversion
// ---------------------------------------------------------------------------
__global__ __launch_bounds__(256) void cvt_f16_kernel(const float* __restrict__ in,
                                                      _Float16* __restrict__ out,
                                                      int n) {
    int i = blockIdx.x * 256 + threadIdx.x;
    if (i < n) out[i] = (_Float16)in[i];
}

// ---------------------------------------------------------------------------
// f16 GEMM: C(MxN) = A(MxK) * W(KxN), row-major. 128x64 tile / 256 threads.
// 8 waves as 4(M)x2(N); each wave owns a 2x2 grid of 16x16 WMMA tiles.
// A tile filled via async global->LDS; B tile transposed through VGPRs.
// ---------------------------------------------------------------------------
#define TM 128
#define TN 64
#define TK 32

// LDS arena (single variable => allocated at LDS byte offset 0):
//   [0, TM*TK)           : sA   (128x32 f16, 8 KB)
//   [TM*TK, +TN*TK)      : sBT  (64x32 f16 transposed W tile, 4 KB)
#define SMEM_ELTS (TM * TK + TN * TK)

template <bool BIAS_F32_OUT>
__global__ __launch_bounds__(256) void gemm_f16_t(
    const _Float16* __restrict__ A, const _Float16* __restrict__ W,
    const float* __restrict__ bias, _Float16* __restrict__ Ch,
    float* __restrict__ Cf, int M, int N, int Kd) {
    __shared__ _Float16 smem[SMEM_ELTS];
    _Float16* sA  = smem;
    _Float16* sBT = smem + TM * TK;

    int n0 = blockIdx.x * TN, m0 = blockIdx.y * TM;
    int t = threadIdx.x, wave = t >> 5, lane = t & 31;
    int mt = (wave & 3) * 32;              // wave's 2 m-tiles: mt, mt+16
    int nt = (wave >> 2) * 32;             // wave's 2 n-tiles: nt, nt+16
    v8f acc00 = {}, acc01 = {}, acc10 = {}, acc11 = {};

    int ar = t >> 2, ac = (t & 3) * 8;     // A fill: 16B per lane, 2 chunks
    unsigned ldsA = (unsigned)((ar * TK + ac) * 2);
    int bk = t >> 3, bn = (t & 7) * 8;     // B fill: 8 cols of one k-row

    for (int kb = 0; kb < Kd; kb += TK) {
        // --- A tile: 128 rows x 32 halves via async global->LDS (8 KB) ---
        async_copy_b128(ldsA,
                        A + (size_t)(m0 + ar) * Kd + kb + ac);
        async_copy_b128(ldsA + (unsigned)(64 * TK * 2),
                        A + (size_t)(m0 + 64 + ar) * Kd + kb + ac);
        // --- B tile: transpose W[kb..kb+31][n0..n0+63] into sBT ---
        {
            v8h w = *(const v8h*)(W + (size_t)(kb + bk) * N + n0 + bn);
#pragma unroll
            for (int j = 0; j < 8; ++j) sBT[(bn + j) * TK + bk] = w[j];
            if (kb + TK < Kd)
                __builtin_prefetch(W + (size_t)(kb + TK + bk) * N + n0 + bn, 0, 3);
        }
        wait_async0();
        __syncthreads();

        v16h a0 = frag_ld(sA,  TK, mt, 0);
        v16h a1 = frag_ld(sA,  TK, mt + 16, 0);
        v16h b0 = frag_ld(sBT, TK, nt, 0);
        v16h b1 = frag_ld(sBT, TK, nt + 16, 0);
        acc00 = wmma_f16(a0, b0, acc00);
        acc01 = wmma_f16(a0, b1, acc01);
        acc10 = wmma_f16(a1, b0, acc10);
        acc11 = wmma_f16(a1, b1, acc11);
        __syncthreads();
    }

    // C/D layout: lanes 0-15 N=lane rows m..m+7; lanes 16-31 N=lane-16, m+8..m+15
    int col  = n0 + nt + (lane & 15);
    int row0 = m0 + mt + ((lane >> 4) ? 8 : 0);
    if (BIAS_F32_OUT) {
        float bi0 = bias[col], bi1 = bias[col + 16];
#pragma unroll
        for (int r = 0; r < 8; ++r) {
            Cf[(size_t)(row0 + r) * N + col]           = acc00[r] + bi0;
            Cf[(size_t)(row0 + r) * N + col + 16]      = acc01[r] + bi1;
            Cf[(size_t)(row0 + 16 + r) * N + col]      = acc10[r] + bi0;
            Cf[(size_t)(row0 + 16 + r) * N + col + 16] = acc11[r] + bi1;
        }
    } else {
#pragma unroll
        for (int r = 0; r < 8; ++r) {
            Ch[(size_t)(row0 + r) * N + col]           = (_Float16)acc00[r];
            Ch[(size_t)(row0 + r) * N + col + 16]      = (_Float16)acc01[r];
            Ch[(size_t)(row0 + 16 + r) * N + col]      = (_Float16)acc10[r];
            Ch[(size_t)(row0 + 16 + r) * N + col + 16] = (_Float16)acc11[r];
        }
    }
}

// ---------------------------------------------------------------------------
// V-projection transpose: (b, k, h*64+d) f16 -> (b, h, d, k) f16
// ---------------------------------------------------------------------------
__global__ __launch_bounds__(256) void vtrans_kernel(const _Float16* __restrict__ vp,
                                                     _Float16* __restrict__ vT) {
    int i = blockIdx.x * 256 + threadIdx.x;       // over B*K*HD = 4M
    int hd = i & (HD - 1);
    int kq = (i >> 10) & (KK - 1);
    int b  = i >> 20;
    int h  = hd >> 6, d = hd & 63;
    vT[(((size_t)(b * HH + h) * DD) + d) * KK + kq] = vp[i];
}

// ---------------------------------------------------------------------------
// Fused attention: per block = one (b,h) x 16 query rows.
//   scores (WMMA) -> LDS -> softmax -> mix supplied -> renorm -> attn@V (WMMA)
// ---------------------------------------------------------------------------
__global__ __launch_bounds__(256) void attn_kernel(
    const _Float16* __restrict__ qp, const _Float16* __restrict__ kp,
    const _Float16* __restrict__ vT, const float* __restrict__ amask,
    const float* __restrict__ supplied, _Float16* __restrict__ outp) {
    __shared__ float sS[16][KK];                  // 64 KB score/attn rows
    int q0 = blockIdx.x * 16;
    int bh = blockIdx.y;
    int b = bh >> 4, h = bh & 15;
    int t = threadIdx.x, wave = t >> 5, lane = t & 31;

    const _Float16* qbase = qp + (size_t)b * QQ * HD + h * DD;
    const _Float16* kbase = kp + (size_t)b * KK * HD + h * DD;

    // ---- Phase 1: scores = (q @ k^T) * SCALE, causal mask, + attention_mask
    for (int ntile = 0; ntile < 8; ++ntile) {
        int kc0 = wave * 128 + ntile * 16;        // key-column tile base
        v8f acc = {};
#pragma unroll
        for (int kk = 0; kk < DD; kk += 32) {
            v16h a = frag_ld(qbase, HD, q0, kk);  // q rows
            v16h f = frag_ld(kbase, HD, kc0, kk); // k rows == columns of k^T
            acc = wmma_f16(a, f, acc);
        }
        int col = kc0 + (lane & 15);
        int r0  = (lane >> 4) ? 8 : 0;
#pragma unroll
        for (int r = 0; r < 8; ++r) {
            int row  = r0 + r;
            int qrow = q0 + row;
            float s = acc[r] * SCALE;
            if (col > qrow) s = NEGV;
            s += amask[((size_t)b * QQ + qrow) * KK + col];
            sS[row][col] = s;
        }
    }
    __syncthreads();

    // ---- Phase 2: softmax + mix with supplied_attn + renormalize
    {
        int row = t >> 4, g = t & 15;             // 16 lanes per row (in-wave)
        int qrow = q0 + row;
        float mx = -3.0e38f;
        for (int k = g; k < KK; k += 16) mx = fmaxf(mx, sS[row][k]);
#pragma unroll
        for (int m = 1; m < 16; m <<= 1) mx = fmaxf(mx, __shfl_xor(mx, m, 16));
        float se = 0.f;
        for (int k = g; k < KK; k += 16) se += __expf(sS[row][k] - mx);
#pragma unroll
        for (int m = 1; m < 16; m <<= 1) se += __shfl_xor(se, m, 16);
        float inv = 1.0f / se;
        const float* sup = supplied + ((size_t)bh * QQ + qrow) * KK;
        float tot = 0.f;
        for (int k = g; k < KK; k += 16) {
            float e    = __expf(sS[row][k] - mx) * inv;
            float mixv = 0.5f * e + 0.5f * sup[k];
            tot += mixv;                          // sum BEFORE causal zeroing
            sS[row][k] = (k > qrow) ? 0.0f : mixv;
        }
#pragma unroll
        for (int m = 1; m < 16; m <<= 1) tot += __shfl_xor(tot, m, 16);
        float rn = 1.0f / (tot + 1e-9f);
        for (int k = g; k < KK; k += 16) sS[row][k] *= rn;
    }
    __syncthreads();

    // ---- Phase 3: out = attn @ v  (waves 0..3 each own a 16-wide d-tile)
    if (wave < 4) {
        int d0 = wave * 16;
        const _Float16* vbase = vT + (size_t)bh * DD * KK;   // (d, k) row-major
        v8f acc = {};
        for (int kc = 0; kc < KK; kc += 32) {
            // A fragment from LDS attn rows, f32 -> f16 on the fly
            int rr = lane & 15;
            int ks = kc + ((lane >> 4) ? 8 : 0);
            const float* prow = &sS[rr][ks];
            v16h a;
#pragma unroll
            for (int i = 0; i < 8; ++i) {
                a[i]     = (_Float16)prow[i];
                a[8 + i] = (_Float16)prow[16 + i];
            }
            v16h f = frag_ld(vbase, KK, d0, kc); // rows of vT == columns of v
            acc = wmma_f16(a, f, acc);
        }
        int col = d0 + (lane & 15);
        int r0  = (lane >> 4) ? 8 : 0;
        _Float16* op = outp + ((size_t)b * QQ + q0) * HD + h * DD;
#pragma unroll
        for (int r = 0; r < 8; ++r)
            op[(size_t)(r0 + r) * HD + col] = (_Float16)acc[r];
    }
}

// ---------------------------------------------------------------------------
// Host launcher
// ---------------------------------------------------------------------------
extern "C" void kernel_launch(void* const* d_in, const int* in_sizes, int n_in,
                              void* d_out, int out_size, void* d_ws, size_t ws_size,
                              hipStream_t stream) {
    const float* query = (const float*)d_in[0];
    const float* keyt  = (const float*)d_in[1];
    const float* value = (const float*)d_in[2];
    const float* amask = (const float*)d_in[3];
    const float* sup   = (const float*)d_in[4];
    const float* Wq    = (const float*)d_in[5];
    const float* Wk    = (const float*)d_in[6];
    const float* Wv    = (const float*)d_in[7];
    const float* Wo    = (const float*)d_in[8];
    const float* bo    = (const float*)d_in[9];
    // d_in[10] = causal_mask: computed analytically in-kernel.

    const size_t MB = 1u << 20;
    char* ws = (char*)d_ws;
    _Float16* q16in   = (_Float16*)(ws + 0 * MB);   // B*Q*DIM f16 = 8 MB
    _Float16* k16in   = (_Float16*)(ws + 8 * MB);
    _Float16* v16in   = (_Float16*)(ws + 16 * MB);
    _Float16* Wq16    = (_Float16*)(ws + 24 * MB);  // 2 MB each
    _Float16* Wk16    = (_Float16*)(ws + 26 * MB);
    _Float16* Wv16    = (_Float16*)(ws + 28 * MB);
    _Float16* Wo16    = (_Float16*)(ws + 30 * MB);
    _Float16* qproj   = (_Float16*)(ws + 32 * MB);  // 8 MB each
    _Float16* kproj   = (_Float16*)(ws + 40 * MB);
    _Float16* vproj   = (_Float16*)(ws + 48 * MB);
    _Float16* vT      = (_Float16*)(ws + 56 * MB);
    _Float16* attnout = (_Float16*)(ws + 64 * MB);

    const int nAct = BB * QQ * DIMM;   // 4,194,304
    const int nW   = DIMM * HD;        // 1,048,576

    // 1) fp32 -> f16 conversions
    cvt_f16_kernel<<<nAct / 256, 256, 0, stream>>>(query, q16in, nAct);
    cvt_f16_kernel<<<nAct / 256, 256, 0, stream>>>(keyt,  k16in, nAct);
    cvt_f16_kernel<<<nAct / 256, 256, 0, stream>>>(value, v16in, nAct);
    cvt_f16_kernel<<<nW / 256, 256, 0, stream>>>(Wq, Wq16, nW);
    cvt_f16_kernel<<<nW / 256, 256, 0, stream>>>(Wk, Wk16, nW);
    cvt_f16_kernel<<<nW / 256, 256, 0, stream>>>(Wv, Wv16, nW);
    cvt_f16_kernel<<<nW / 256, 256, 0, stream>>>(Wo, Wo16, nW);

    // 2) Q/K/V projections: (B*Q x DIM) @ (DIM x HD)
    dim3 gg(HD / TN, (BB * QQ) / TM);
    gemm_f16_t<false><<<gg, 256, 0, stream>>>(q16in, Wq16, nullptr, qproj, nullptr,
                                              BB * QQ, HD, DIMM);
    gemm_f16_t<false><<<gg, 256, 0, stream>>>(k16in, Wk16, nullptr, kproj, nullptr,
                                              BB * QQ, HD, DIMM);
    gemm_f16_t<false><<<gg, 256, 0, stream>>>(v16in, Wv16, nullptr, vproj, nullptr,
                                              BB * QQ, HD, DIMM);

    // 3) transpose V projection to (b,h,d,k)
    vtrans_kernel<<<nAct / 256, 256, 0, stream>>>(vproj, vT);

    // 4) fused attention
    dim3 ga(QQ / 16, BB * HH);
    attn_kernel<<<ga, 256, 0, stream>>>(qproj, kproj, vT, amask, sup, attnout);

    // 5) output projection + bias -> fp32 d_out
    gemm_f16_t<true><<<gg, 256, 0, stream>>>(attnout, Wo16, bo, nullptr,
                                             (float*)d_out, BB * QQ, HD, DIMM);
}
